// Generator_z2seq_29609504539214
// MI455X (gfx1250) — compile-verified, matrix operand order checked by reading, not run on hardware
//
#include <hip/hip_runtime.h>

// ---------------------------------------------------------------------------
// Types / helpers
// ---------------------------------------------------------------------------
typedef __attribute__((ext_vector_type(16))) __bf16 v16bf;
typedef __attribute__((ext_vector_type(8)))  float  v8f;
typedef __attribute__((ext_vector_type(4)))  unsigned u32x4;
typedef __attribute__((ext_vector_type(2)))  unsigned u32x2;
typedef __attribute__((ext_vector_type(4)))  float    f32x4;

union Frag32 { v16bf v; u32x4 q[2]; };
union Half8  { u32x4 q; __bf16 h[8]; };
union Half4  { u32x2 q; __bf16 h[4]; };

__device__ __forceinline__ __bf16 f2bf(float f) {
  unsigned u = __float_as_uint(f);
  unsigned r = (u + 0x7FFFu + ((u >> 16) & 1u)) >> 16;   // round-to-nearest-even
  unsigned short s = (unsigned short)r;
  return __builtin_bit_cast(__bf16, s);
}
__device__ __forceinline__ float bf2f(__bf16 b) {
  unsigned short s = __builtin_bit_cast(unsigned short, b);
  return __uint_as_float(((unsigned)s) << 16);
}
__device__ __forceinline__ float sigmoidf_(float x) { return 1.0f / (1.0f + __expf(-x)); }
__device__ __forceinline__ v8f zerov8() { v8f z = {0.f,0.f,0.f,0.f,0.f,0.f,0.f,0.f}; return z; }

#define WMMA_BF16(a, b, c) \
  __builtin_amdgcn_wmma_f32_16x16x32_bf16(false, (a), false, (b), (short)0, (c), false, false)

// A fragment (16x32 bf16) from a swizzled row-major LDS tile.
// Lane L: m = m0 + (L&15), hi = L>>4; elements 0..7 hold K = k0+hi*8+0..7,
// elements 8..15 hold K = k0+16+hi*8+0..7  (per CDNA5 ISA 16-bit A layout).
__device__ __forceinline__ v16bf ldsA(const __bf16* base, int rowHalves, int m0, int k0) {
  int lane = threadIdx.x & 31;
  int m = m0 + (lane & 15);
  int hi = lane >> 4;
  const char* row = (const char*)base + (size_t)m * rowHalves * 2;
  int sw = (m & 15) << 4;
  Frag32 f;
  f.q[0] = *(const u32x4*)(row + (((k0 + hi * 8) * 2) ^ sw));
  f.q[1] = *(const u32x4*)(row + (((k0 + hi * 8 + 16) * 2) ^ sw));
  return f.v;
}

__device__ __forceinline__ void ldA4(v16bf* a, const __bf16* base, int rowHalves, int k0) {
  a[0] = ldsA(base, rowHalves, 0,  k0);
  a[1] = ldsA(base, rowHalves, 16, k0);
  a[2] = ldsA(base, rowHalves, 32, k0);
  a[3] = ldsA(base, rowHalves, 48, k0);
}

// B fragment (32x16 bf16) from pre-packed global weights: 1KB per fragment,
// lane L owns 32 contiguous bytes -> two global b128 loads, fully coalesced.
__device__ __forceinline__ v16bf gB(const char* Wp, int ntiles, int kt, int nt) {
  int lane = threadIdx.x & 31;
  const char* p = Wp + (((size_t)(kt * ntiles + nt)) << 10) + (lane << 5);
  Frag32 f;
  f.q[0] = *(const u32x4*)p;
  f.q[1] = *(const u32x4*)(p + 16);
  return f.v;
}

// ---------------------------------------------------------------------------
// Kernel 1: pack an (K,N) f32 weight into bf16 WMMA-B-fragment layout
// ---------------------------------------------------------------------------
__global__ __launch_bounds__(256) void pack_w(const float* __restrict__ W,
                                              char* __restrict__ Wp, int K, int N) {
  int lane = threadIdx.x & 31;
  int frag = blockIdx.x * (blockDim.x >> 5) + (threadIdx.x >> 5);
  int ntiles = N >> 4;
  int total = (K >> 5) * ntiles;
  if (frag >= total) return;
  int kt = frag / ntiles, nt = frag % ntiles;
  int n = (nt << 4) + (lane & 15);
  int hi = lane >> 4;
  union { __bf16 h[16]; u32x4 q[2]; } buf;
#pragma unroll
  for (int i = 0; i < 16; ++i) {
    int k = (kt << 5) + (i < 8 ? hi * 8 + i : 16 + hi * 8 + (i - 8));
    buf.h[i] = f2bf(W[(size_t)k * N + n]);
  }
  char* dst = Wp + ((size_t)frag << 10) + (lane << 5);
  *(u32x4*)dst = buf.q[0];
  *(u32x4*)(dst + 16) = buf.q[1];
}

// ---------------------------------------------------------------------------
// Kernel 2: x = z @ W_z + b_z   (4096x512, K=128; tiny, VALU f32 -> bf16)
// ---------------------------------------------------------------------------
__global__ __launch_bounds__(256) void x_gemm(const float* __restrict__ z,
                                              const float* __restrict__ Wz,
                                              const float* __restrict__ bz,
                                              __bf16* __restrict__ xout) {
  int idx = blockIdx.x * blockDim.x + threadIdx.x;   // 4096*512
  int m = idx >> 9, n = idx & 511;
  float acc = bz[n];
  const float* zr = z + (size_t)m * 128;
#pragma unroll 4
  for (int k = 0; k < 128; ++k) acc = fmaf(zr[k], Wz[(size_t)k * 512 + n], acc);
  xout[idx] = f2bf(acc);
}

// ---------------------------------------------------------------------------
// Kernel 3: belta = exp(-relu(time_delta @ W_belta + b))  via WMMA bf16
// grid = (B/16)*S workgroups; each: 16 rows x 1024 cols, K=512
// ---------------------------------------------------------------------------
__global__ __launch_bounds__(256) void belta_k(const float* __restrict__ td,
                                               const char* __restrict__ Wbp,
                                               const float* __restrict__ bb,
                                               __bf16* __restrict__ belta) {
  int bt = blockIdx.x / 12, s = blockIdx.x % 12;
  int b0 = bt * 16;
  extern __shared__ char smem[];
  __bf16* sA = (__bf16*)smem;                      // 16 x 512 halves, swizzled
  int t = threadIdx.x;
  // stage A tile: f32 -> bf16
#pragma unroll
  for (int it = 0; it < 8; ++it) {
    int c = t + it * 256;                          // 2048 chunks of 4 f32
    int r = c >> 7;
    int k4 = (c & 127) * 4;
    f32x4 v = *(const f32x4*)(td + (((size_t)(b0 + r) * 12 + s) * 512 + k4));
    Half4 p;
    p.h[0] = f2bf(v[0]); p.h[1] = f2bf(v[1]); p.h[2] = f2bf(v[2]); p.h[3] = f2bf(v[3]);
    *(u32x2*)((char*)sA + r * 1024 + (((k4 * 2)) ^ ((r & 15) << 4))) = p.q;
  }
  __syncthreads();

  int w = t >> 5, lane = t & 31, n = lane & 15, hi = lane >> 4;
  for (int g = 0; g < 2; ++g) {
    int ntb = w * 8 + g * 4;
    v8f acc[4];
#pragma unroll
    for (int j = 0; j < 4; ++j) acc[j] = zerov8();
    for (int kt = 0; kt < 16; ++kt) {
      v16bf a = ldsA(sA, 512, 0, kt * 32);
#pragma unroll
      for (int j = 0; j < 4; ++j) {
        v16bf b = gB(Wbp, 64, kt, ntb + j);
        acc[j] = WMMA_BF16(a, b, acc[j]);
      }
    }
#pragma unroll
    for (int j = 0; j < 4; ++j) {
      int ncol = (ntb + j) * 16 + n;
      float bias = bb[ncol];
#pragma unroll
      for (int i = 0; i < 8; ++i) {
        float vv = acc[j][i] + bias;
        vv = __expf(-fmaxf(vv, 0.0f));
        int row = b0 + i + 8 * hi;
        belta[((size_t)row * 12 + s) * 1024 + ncol] = f2bf(vv);
      }
    }
  }
}

// ---------------------------------------------------------------------------
// Kernel 4: persistent fused GRU recurrence (12 steps, batch tile of 64 rows)
// 512 threads = 16 waves (4 waves per SIMD32, one workgroup fills the WGP).
// LDS (dynamic, 320KB): sSt 64x1024 bf16 | sRS 64x1024 bf16 | sX 64x512 bf16
// ---------------------------------------------------------------------------
__global__ __launch_bounds__(512) void rnn_k(const __bf16* __restrict__ xin,
                                             const __bf16* __restrict__ belta,
                                             const char* __restrict__ W1p,
                                             const char* __restrict__ W2p,
                                             const char* __restrict__ W3p,
                                             const char* __restrict__ Wop,
                                             const float* __restrict__ b1,
                                             const float* __restrict__ b2,
                                             const float* __restrict__ b3,
                                             const float* __restrict__ bo,
                                             char* __restrict__ u_ws,
                                             float* __restrict__ out) {
  extern __shared__ char smem[];
  __bf16* sSt = (__bf16*)smem;                       // state / state_pre
  __bf16* sRS = sSt + 64 * 1024;                     // r * state_pre
  __bf16* sX  = sRS + 64 * 1024;                     // current step input
  int b0 = blockIdx.x * 64;
  int t = threadIdx.x, lane = t & 31, w = t >> 5;    // w in [0,16)
  int n = lane & 15, hi = lane >> 4;

  // stage x tile (bf16, swizzled) and zero state
#pragma unroll
  for (int it = 0; it < 8; ++it) {
    int c = t + it * 512;                            // 4096 chunks of 8 halves
    int r = c >> 6;
    int h8 = (c & 63) * 8;
    u32x4 v = *(const u32x4*)(xin + ((size_t)(b0 + r) * 512 + h8));
    *(u32x4*)((char*)sX + r * 1024 + ((h8 * 2) ^ ((r & 15) << 4))) = v;
  }
  u32x4 z4 = {0u, 0u, 0u, 0u};
#pragma unroll
  for (int it = 0; it < 16; ++it) {
    int c = t + it * 512;                            // 8192 chunks
    int r = c >> 7;
    int h8 = (c & 127) * 8;
    *(u32x4*)((char*)sSt + r * 2048 + ((h8 * 2) ^ ((r & 15) << 4))) = z4;
  }
  __syncthreads();

  for (int s = 0; s < 12; ++s) {
    // ---- P1: state_pre = belta_s * state (in place) ----
    for (int it = 0; it < 16; ++it) {
      int c = t + it * 512;
      int r = c >> 7;
      int h8 = (c & 127) * 8;
      char* ps = (char*)sSt + r * 2048 + ((h8 * 2) ^ ((r & 15) << 4));
      Half8 a; a.q = *(u32x4*)ps;
      Half8 bq; bq.q = *(const u32x4*)(belta + (((size_t)(b0 + r) * 12 + s) * 1024 + h8));
#pragma unroll
      for (int i = 0; i < 8; ++i) a.h[i] = f2bf(bf2f(a.h[i]) * bf2f(bq.h[i]));
      *(u32x4*)ps = a.q;
    }
    __syncthreads();

    // ---- P2: fused u/r GEMMs over cat=[state_pre | x], K=1536 ----
    for (int g = 0; g < 4; ++g) {
      int ht = w + 16 * g;                           // 64 h-tiles over 16 waves
      v8f au[4], ar[4];
#pragma unroll
      for (int j = 0; j < 4; ++j) { au[j] = zerov8(); ar[j] = zerov8(); }
      for (int kt = 0; kt < 48; ++kt) {
        v16bf a[4];
        if (kt < 32) ldA4(a, sSt, 1024, kt * 32);
        else         ldA4(a, sX,  512,  kt * 32 - 1024);
        v16bf bu = gB(W1p, 64, kt, ht);
        v16bf br = gB(W2p, 64, kt, ht);
#pragma unroll
        for (int mi = 0; mi < 4; ++mi) {
          au[mi] = WMMA_BF16(a[mi], bu, au[mi]);
          ar[mi] = WMMA_BF16(a[mi], br, ar[mi]);
        }
      }
      float biasu = b1[ht * 16 + n];
      float biasr = b2[ht * 16 + n];
#pragma unroll
      for (int mi = 0; mi < 4; ++mi) {
        Half8 uh;
#pragma unroll
        for (int i = 0; i < 8; ++i) uh.h[i] = f2bf(sigmoidf_(au[mi][i] + biasu));
        size_t uoff = ((((size_t)(blockIdx.x * 4 + mi) * 64 + ht) * 32 + lane) << 4);
        *(u32x4*)(u_ws + uoff) = uh.q;               // same wave reads it back in P3
#pragma unroll
        for (int i = 0; i < 8; ++i) {
          float rv = sigmoidf_(ar[mi][i] + biasr);
          int row = mi * 16 + i + 8 * hi;
          int col = ht * 16 + n;
          int so = ((col * 2) ^ ((row & 15) << 4));
          float spv = bf2f(*(__bf16*)((char*)sSt + row * 2048 + so));
          *(__bf16*)((char*)sRS + row * 2048 + so) = f2bf(rv * spv);
        }
      }
    }
    __syncthreads();

    // ---- P3: new_state = tanh([r*state | x] @ W3 + b3); blend with u ----
    for (int g = 0; g < 4; ++g) {
      int ht = w + 16 * g;
      v8f ac[4];
#pragma unroll
      for (int j = 0; j < 4; ++j) ac[j] = zerov8();
      for (int kt = 0; kt < 48; ++kt) {
        v16bf a[4];
        if (kt < 32) ldA4(a, sRS, 1024, kt * 32);
        else         ldA4(a, sX,  512,  kt * 32 - 1024);
        v16bf b = gB(W3p, 64, kt, ht);
#pragma unroll
        for (int mi = 0; mi < 4; ++mi) ac[mi] = WMMA_BF16(a[mi], b, ac[mi]);
      }
      float bias = b3[ht * 16 + n];
#pragma unroll
      for (int mi = 0; mi < 4; ++mi) {
        size_t uoff = ((((size_t)(blockIdx.x * 4 + mi) * 64 + ht) * 32 + lane) << 4);
        Half8 uh; uh.q = *(const u32x4*)(u_ws + uoff);
#pragma unroll
        for (int i = 0; i < 8; ++i) {
          float tv = tanhf(ac[mi][i] + bias);
          int row = mi * 16 + i + 8 * hi;
          int col = ht * 16 + n;
          int so = ((col * 2) ^ ((row & 15) << 4));
          char* sp = (char*)sSt + row * 2048 + so;
          float spv = bf2f(*(__bf16*)sp);
          float uv = bf2f(uh.h[i]);
          *(__bf16*)sp = f2bf((1.0f - uv) * spv + uv * tv);   // new state
        }
      }
    }
    __syncthreads();

    // prefetch next step's belta tile (HBM stream) while P4 runs
    if (s < 11) {
#pragma unroll
      for (int it = 0; it < 4; ++it) {
        int L = t + it * 512;                        // 2048 cachelines of 64B
        int r = L >> 5;
        const char* pf = (const char*)belta +
            (((size_t)(b0 + r) * 12 + (s + 1)) * 1024) * 2 + (size_t)(L & 31) * 64;
        __builtin_prefetch(pf, 0, 1);
      }
    }

    // ---- P4: out = state @ W_out + b_out ; write f32 out, bf16 -> sX ----
    for (int g = 0; g < 2; ++g) {
      int nt = w + 16 * g;                           // 32 n-tiles over 16 waves
      v8f ac[4];
#pragma unroll
      for (int j = 0; j < 4; ++j) ac[j] = zerov8();
      for (int kt = 0; kt < 32; ++kt) {
        v16bf a[4];
        ldA4(a, sSt, 1024, kt * 32);
        v16bf b = gB(Wop, 32, kt, nt);
#pragma unroll
        for (int mi = 0; mi < 4; ++mi) ac[mi] = WMMA_BF16(a[mi], b, ac[mi]);
      }
      float bias = bo[nt * 16 + n];
#pragma unroll
      for (int mi = 0; mi < 4; ++mi) {
#pragma unroll
        for (int i = 0; i < 8; ++i) {
          float ov = ac[mi][i] + bias;
          int row = mi * 16 + i + 8 * hi;
          int col = nt * 16 + n;
          out[((size_t)(b0 + row) * 12 + s) * 512 + col] = ov;
          *(__bf16*)((char*)sX + row * 1024 + ((col * 2) ^ ((row & 15) << 4))) = f2bf(ov);
        }
      }
    }
    __syncthreads();
  }
}

// ---------------------------------------------------------------------------
// Host launcher
// ---------------------------------------------------------------------------
extern "C" void kernel_launch(void* const* d_in, const int* in_sizes, int n_in,
                              void* d_out, int out_size, void* d_ws, size_t ws_size,
                              hipStream_t stream) {
  const float* z   = (const float*)d_in[0];
  const float* td  = (const float*)d_in[1];
  const float* Wb  = (const float*)d_in[2];
  const float* bb  = (const float*)d_in[3];
  const float* Wz  = (const float*)d_in[4];
  const float* bz  = (const float*)d_in[5];
  const float* W1  = (const float*)d_in[6];
  const float* b1  = (const float*)d_in[7];
  const float* W2  = (const float*)d_in[8];
  const float* b2  = (const float*)d_in[9];
  const float* W3  = (const float*)d_in[10];
  const float* b3  = (const float*)d_in[11];
  const float* Wo  = (const float*)d_in[12];
  const float* bo  = (const float*)d_in[13];

  char* ws = (char*)d_ws;
  size_t off = 0;
  char* W1p = ws + off; off += (size_t)48 * 64 * 1024;        // 3MB
  char* W2p = ws + off; off += (size_t)48 * 64 * 1024;        // 3MB
  char* W3p = ws + off; off += (size_t)48 * 64 * 1024;        // 3MB
  char* Wop = ws + off; off += (size_t)32 * 32 * 1024;        // 1MB
  char* Wbp = ws + off; off += (size_t)16 * 64 * 1024;        // 1MB
  __bf16* x_bf = (__bf16*)(ws + off); off += (size_t)4096 * 512 * 2;   // 4MB
  char* u_ws = ws + off; off += (size_t)256 * 64 * 32 * 16;   // 8MB
  __bf16* belta = (__bf16*)(ws + off);                         // 100MB

  // pack weights into WMMA B-fragment layout (bf16)
  pack_w<<<(3072 + 7) / 8, 256, 0, stream>>>(W1, W1p, 1536, 1024);
  pack_w<<<(3072 + 7) / 8, 256, 0, stream>>>(W2, W2p, 1536, 1024);
  pack_w<<<(3072 + 7) / 8, 256, 0, stream>>>(W3, W3p, 1536, 1024);
  pack_w<<<(1024 + 7) / 8, 256, 0, stream>>>(Wo, Wop, 1024, 512);
  pack_w<<<(1024 + 7) / 8, 256, 0, stream>>>(Wb, Wbp, 512, 1024);

  // x = z @ W_z + b_z
  x_gemm<<<(4096 * 512) / 256, 256, 0, stream>>>(z, Wz, bz, x_bf);

  // belta = exp(-relu(time_delta @ W_belta + b_belta))
  belta_k<<<(4096 / 16) * 12, 256, 16 * 1024, stream>>>(td, Wbp, bb, belta);

  // persistent fused GRU recurrence, 320KB LDS, 16 waves per workgroup
  rnn_k<<<64, 512, 320 * 1024, stream>>>(x_bf, belta, W1p, W2p, W3p, Wop,
                                         b1, b2, b3, bo, u_ws, (float*)d_out);
}
